// RNNDecoder_34497177321888
// MI455X (gfx1250) — matrix operand order repaired and checked by reading, not compiled, not run on hardware
//
#include <hip/hip_runtime.h>
#include <hip/hip_bf16.h>

// ---------------------------------------------------------------------------
// RNN decoder (LSTM + dot attention + vocab projection) for MI455X / gfx1250.
// Dominant GEMMs run on the bf16 matrix pipe (V_WMMA_F32_16X16X32_BF16) using
// a hi/lo split (3x bf16 ~= fp32) for near-f32 accuracy at 8x the K-per-issue
// of the f32 path. The one-time input-gate GEMM stays on V_WMMA_F32_16X16X4_F32.
// ---------------------------------------------------------------------------

typedef __attribute__((ext_vector_type(2)))  float  v2f;
typedef __attribute__((ext_vector_type(8)))  float  v8f;
typedef __attribute__((ext_vector_type(8)))  __bf16 v8bf;
typedef __attribute__((ext_vector_type(16))) __bf16 v16bf;

#define B   32
#define TD  64      // T_dec
#define TE  128     // T_enc
#define E   256
#define H   512
#define G4  2048    // 4*H
#define V   32000

// ---------------------------------------------------------------------------
// f32 WMMA tile (16x16, K stepped by 4) : C = A(MxK) * W^T, W row-major [N,K]
// ---------------------------------------------------------------------------
__device__ __forceinline__ v8f wmma_tile_f32(const float* __restrict__ A, int lda,
                                             const float* __restrict__ W, int ldw,
                                             int m0, int n0, int K, v8f acc) {
  const int lane = threadIdx.x & 31;
  const int half = lane >> 4;
  const int lr   = lane & 15;
  const float* __restrict__ arow = A + (size_t)(m0 + lr) * lda + 2 * half;
  const float* __restrict__ wrow = W + (size_t)(n0 + lr) * ldw + 2 * half;
#pragma unroll 8
  for (int k = 0; k < K; k += 4) {
    v2f a, b;
    a.x = arow[k];  a.y = arow[k + 1];
    b.x = wrow[k];  b.y = wrow[k + 1];
    acc = __builtin_amdgcn_wmma_f32_16x16x4_f32(false, a, false, b,
                                                (short)0, acc, false, false);
  }
  return acc;
}

// ---------------------------------------------------------------------------
// bf16 split WMMA tile (16x16, K stepped by 32, 3 WMMAs per step).
// A fragment (16-bit A 16x32, ISA 7.12.2): lane half h holds row M=lane%16,
//   elems [0..15] = K[k+16h .. k+16h+15] contiguous (same k order as B).
// B fragment: lane half h holds col N=lane%16, K[k+16h .. k+16h+15] contiguous.
// ---------------------------------------------------------------------------
__device__ __forceinline__ v16bf load_fragA(const __bf16* __restrict__ p) {
  v8bf x = *(const v8bf*)p;           // 16B
  v8bf y = *(const v8bf*)(p + 8);     // 16B
  return __builtin_shufflevector(x, y, 0, 1, 2, 3, 4, 5, 6, 7,
                                       8, 9, 10, 11, 12, 13, 14, 15);
}

__device__ __forceinline__ v8f wmma_tile_bf16s(
    const __bf16* __restrict__ Ahi, const __bf16* __restrict__ Alo, int lda,
    const __bf16* __restrict__ Whi, const __bf16* __restrict__ Wlo, int ldw,
    int m0, int n0, int K, v8f acc) {
  const int lane = threadIdx.x & 31;
  const int half = lane >> 4;
  const int lr   = lane & 15;
  const __bf16* __restrict__ ah = Ahi + (size_t)(m0 + lr) * lda + 16 * half;
  const __bf16* __restrict__ al = Alo + (size_t)(m0 + lr) * lda + 16 * half;
  const __bf16* __restrict__ wh = Whi + (size_t)(n0 + lr) * ldw + 16 * half;
  const __bf16* __restrict__ wl = Wlo + (size_t)(n0 + lr) * ldw + 16 * half;
#pragma unroll 4
  for (int k = 0; k < K; k += 32) {
    v16bf a_hi = load_fragA(ah + k);
    v16bf a_lo = load_fragA(al + k);
    v16bf b_hi = *(const v16bf*)(wh + k);   // two b128s
    v16bf b_lo = *(const v16bf*)(wl + k);
    // A*B ~= Ahi*Bhi + Ahi*Blo + Alo*Bhi   (Alo*Blo ~ 2^-16 rel, dropped)
    acc = __builtin_amdgcn_wmma_f32_16x16x32_bf16(false, a_hi, false, b_hi,
                                                  (short)0, acc, false, false);
    acc = __builtin_amdgcn_wmma_f32_16x16x32_bf16(false, a_hi, false, b_lo,
                                                  (short)0, acc, false, false);
    acc = __builtin_amdgcn_wmma_f32_16x16x32_bf16(false, a_lo, false, b_hi,
                                                  (short)0, acc, false, false);
  }
  return acc;
}

__device__ __forceinline__ float sigm(float x) { return 1.0f / (1.0f + __expf(-x)); }

__device__ __forceinline__ void split2(float x, __bf16* hi, __bf16* lo) {
  __bf16 h = (__bf16)x;
  *hi = h;
  *lo = (__bf16)(x - (float)h);
}

// ---------------------------------------------------------------------------
// weight split: f32 -> (bf16 hi, bf16 lo)
// ---------------------------------------------------------------------------
__global__ void k_split(const float* __restrict__ src,
                        __bf16* __restrict__ hi, __bf16* __restrict__ lo, int n) {
  int i = blockIdx.x * blockDim.x + threadIdx.x;
  if (i >= n) return;
  split2(src[i], hi + i, lo + i);
}

// ---------------------------------------------------------------------------
// 1) Embedding gather: emb[(b*TD+t), e] = (idx==0) ? 0 : emb_W[idx, e]
// ---------------------------------------------------------------------------
__global__ void k_embed(const long long* __restrict__ dec_X,
                        const float* __restrict__ emb_W,
                        float* __restrict__ emb) {
  int i = blockIdx.x * blockDim.x + threadIdx.x;
  if (i >= B * TD * E) return;
  int row = i >> 8;
  int e   = i & (E - 1);
  long long idx = dec_X[row];
  emb[i] = (idx == 0) ? 0.0f : emb_W[(size_t)idx * E + e];
}

// ---------------------------------------------------------------------------
// 2) Xg = emb @ W_ih^T : [2048, 2048], K=256 (one-time, f32 WMMA)
// ---------------------------------------------------------------------------
__global__ void k_xg(const float* __restrict__ emb,
                     const float* __restrict__ W_ih,
                     float* __restrict__ Xg) {
  int wave = blockIdx.x * (blockDim.x >> 5) + (threadIdx.x >> 5);
  const int tiles_n = G4 / 16;
  int mt = wave / tiles_n;
  int nt = wave % tiles_n;
  if (mt >= (B * TD) / 16) return;
  v8f acc = {};
  acc = wmma_tile_f32(emb, E, W_ih, E, mt * 16, nt * 16, E, acc);
  const int lane = threadIdx.x & 31, half = lane >> 4, lr = lane & 15;
  int n = nt * 16 + lr;
#pragma unroll
  for (int r = 0; r < 8; ++r) {
    int m = mt * 16 + r + 8 * half;
    Xg[(size_t)m * G4 + n] = acc[r];
  }
}

// ---------------------------------------------------------------------------
// 3a) gates: g = h @ W_hh^T + Xg[b*TD+t] + (b_ih+b_hh)  : [32,2048], K=512
//     h is read (hi/lo) from the left half of the hcat buffers (ld = 2H).
// ---------------------------------------------------------------------------
__global__ void k_gates(const float* __restrict__ Xg,
                        const __bf16* __restrict__ hcat_hi,
                        const __bf16* __restrict__ hcat_lo,
                        const __bf16* __restrict__ Whh_hi,
                        const __bf16* __restrict__ Whh_lo,
                        const float* __restrict__ b_ih,
                        const float* __restrict__ b_hh,
                        float* __restrict__ g, int t) {
  int wave = blockIdx.x * (blockDim.x >> 5) + (threadIdx.x >> 5);
  const int tiles_n = G4 / 16;
  int mt = wave & 1;
  int nt = wave >> 1;
  if (nt >= tiles_n) return;
  v8f acc = {};
  acc = wmma_tile_bf16s(hcat_hi, hcat_lo, 2 * H, Whh_hi, Whh_lo, H,
                        mt * 16, nt * 16, H, acc);
  const int lane = threadIdx.x & 31, half = lane >> 4, lr = lane & 15;
  int n = nt * 16 + lr;
  float bias = b_ih[n] + b_hh[n];
#pragma unroll
  for (int r = 0; r < 8; ++r) {
    int m = mt * 16 + r + 8 * half;
    g[(size_t)m * G4 + n] = acc[r] + Xg[((size_t)m * TD + t) * G4 + n] + bias;
  }
}

// ---------------------------------------------------------------------------
// 3b) LSTM elementwise (torch gate order i,f,g,o). Writes f32 h/c state and
//     the hi/lo h into the left half of the hcat buffers.
// ---------------------------------------------------------------------------
__global__ void k_lstm(const float* __restrict__ g,
                       float* __restrict__ c_state,
                       float* __restrict__ h_state,
                       __bf16* __restrict__ hcat_hi,
                       __bf16* __restrict__ hcat_lo) {
  int i = blockIdx.x * blockDim.x + threadIdx.x;
  if (i >= B * H) return;
  int b = i >> 9;
  int j = i & (H - 1);
  const float* gb = g + (size_t)b * G4;
  float ig = gb[j], fg = gb[H + j], gg = gb[2 * H + j], og = gb[3 * H + j];
  float c = sigm(fg) * c_state[i] + sigm(ig) * tanhf(gg);
  float h = sigm(og) * tanhf(c);
  c_state[i] = c;
  h_state[i] = h;
  size_t o = (size_t)b * (2 * H) + j;
  split2(h, hcat_hi + o, hcat_lo + o);
}

// ---------------------------------------------------------------------------
// 3c) Dot attention per batch row; context written hi/lo into the right half
//     of the hcat buffers. One block (TE threads) per row.
// ---------------------------------------------------------------------------
__global__ void k_attn(const float* __restrict__ enc,
                       const float* __restrict__ h_state,
                       __bf16* __restrict__ hcat_hi,
                       __bf16* __restrict__ hcat_lo) {
  __shared__ float sc[TE];
  int b = blockIdx.x;
  int tid = threadIdx.x;
  const float* hb = h_state + (size_t)b * H;
  const float* e  = enc + ((size_t)b * TE + tid) * H;
  float s = 0.0f;
  for (int k = 0; k < H; ++k) s += e[k] * hb[k];
  sc[tid] = s;
  __syncthreads();
  float m = -1e30f;
  for (int i = 0; i < TE; ++i) m = fmaxf(m, sc[i]);
  float denom = 0.0f;
  for (int i = 0; i < TE; ++i) denom += __expf(sc[i] - m);
  float a = __expf(sc[tid] - m) / denom;
  __syncthreads();
  sc[tid] = a;
  __syncthreads();
  for (int j = tid; j < H; j += TE) {
    float acc = 0.0f;
    for (int t = 0; t < TE; ++t) acc += sc[t] * enc[((size_t)b * TE + t) * H + j];
    size_t o = (size_t)b * (2 * H) + H + j;
    split2(acc, hcat_hi + o, hcat_lo + o);
  }
}

// ---------------------------------------------------------------------------
// 3d) out = tanh(hcat @ attn_W^T + attn_b) : [32,512], K=1024. Writes hi/lo.
// ---------------------------------------------------------------------------
__global__ void k_attnout(const __bf16* __restrict__ hcat_hi,
                          const __bf16* __restrict__ hcat_lo,
                          const __bf16* __restrict__ aW_hi,
                          const __bf16* __restrict__ aW_lo,
                          const float* __restrict__ attn_b,
                          __bf16* __restrict__ out_hi,
                          __bf16* __restrict__ out_lo) {
  int wave = blockIdx.x * (blockDim.x >> 5) + (threadIdx.x >> 5);
  const int tiles_n = H / 16;
  int mt = wave & 1;
  int nt = wave >> 1;
  if (nt >= tiles_n) return;
  v8f acc = {};
  acc = wmma_tile_bf16s(hcat_hi, hcat_lo, 2 * H, aW_hi, aW_lo, 2 * H,
                        mt * 16, nt * 16, 2 * H, acc);
  const int lane = threadIdx.x & 31, half = lane >> 4, lr = lane & 15;
  int n = nt * 16 + lr;
  float bias = attn_b[n];
#pragma unroll
  for (int r = 0; r < 8; ++r) {
    int m = mt * 16 + r + 8 * half;
    float v = tanhf(acc[r] + bias);
    split2(v, out_hi + (size_t)m * H + n, out_lo + (size_t)m * H + n);
  }
}

// ---------------------------------------------------------------------------
// 3e) logits[b,t,:] = out @ fc_W^T + fc_b : [32,32000], K=512 (dominant GEMM)
// ---------------------------------------------------------------------------
__global__ void k_fc(const __bf16* __restrict__ out_hi,
                     const __bf16* __restrict__ out_lo,
                     const __bf16* __restrict__ fcW_hi,
                     const __bf16* __restrict__ fcW_lo,
                     const float* __restrict__ fc_b,
                     float* __restrict__ logits, int t) {
  int wave = blockIdx.x * (blockDim.x >> 5) + (threadIdx.x >> 5);
  const int tiles_n = V / 16;
  int mt = wave & 1;
  int nt = wave >> 1;
  if (nt >= tiles_n) return;
  v8f acc = {};
  acc = wmma_tile_bf16s(out_hi, out_lo, H, fcW_hi, fcW_lo, H,
                        mt * 16, nt * 16, H, acc);
  const int lane = threadIdx.x & 31, half = lane >> 4, lr = lane & 15;
  int n = nt * 16 + lr;
  float bias = fc_b[n];
#pragma unroll
  for (int r = 0; r < 8; ++r) {
    int m = mt * 16 + r + 8 * half;
    logits[((size_t)m * TD + t) * V + n] = acc[r] + bias;
  }
}

// ---------------------------------------------------------------------------
// state init / final copy
// ---------------------------------------------------------------------------
__global__ void k_init(const float* __restrict__ h0, const float* __restrict__ c0,
                       float* __restrict__ h_state, float* __restrict__ c_state,
                       __bf16* __restrict__ hcat_hi, __bf16* __restrict__ hcat_lo) {
  int i = blockIdx.x * blockDim.x + threadIdx.x;
  if (i >= B * H) return;
  float h = h0[i];
  h_state[i] = h;
  c_state[i] = c0[i];
  int b = i >> 9, j = i & (H - 1);
  size_t o = (size_t)b * (2 * H) + j;
  split2(h, hcat_hi + o, hcat_lo + o);
}

__global__ void k_final(const float* __restrict__ h_state,
                        const float* __restrict__ c_state,
                        float* __restrict__ d_out) {
  int i = blockIdx.x * blockDim.x + threadIdx.x;
  if (i >= B * H) return;
  const size_t base = (size_t)B * TD * V;
  d_out[base + i] = h_state[i];
  d_out[base + (size_t)B * H + i] = c_state[i];
}

// ---------------------------------------------------------------------------
extern "C" void kernel_launch(void* const* d_in, const int* in_sizes, int n_in,
                              void* d_out, int out_size, void* d_ws, size_t ws_size,
                              hipStream_t stream) {
  (void)in_sizes; (void)n_in; (void)out_size; (void)ws_size;

  const long long* dec_X = (const long long*)d_in[0];
  const float* enc    = (const float*)d_in[1];
  const float* h0     = (const float*)d_in[2];
  const float* c0     = (const float*)d_in[3];
  const float* emb_W  = (const float*)d_in[4];
  const float* W_ih   = (const float*)d_in[5];
  const float* W_hh   = (const float*)d_in[6];
  const float* b_ih   = (const float*)d_in[7];
  const float* b_hh   = (const float*)d_in[8];
  const float* attn_W = (const float*)d_in[9];
  const float* attn_b = (const float*)d_in[10];
  const float* fc_W   = (const float*)d_in[11];
  const float* fc_b   = (const float*)d_in[12];
  float* outp = (float*)d_out;

  // workspace carve-up (256B-aligned chunks)
  char* p = (char*)d_ws;
  auto carve = [&p](size_t bytes) -> void* {
    void* r = (void*)p;
    p += (bytes + 255) & ~(size_t)255;
    return r;
  };
  float*  emb      = (float*)carve((size_t)B * TD * E * 4);       //  2.0 MB
  float*  Xg       = (float*)carve((size_t)B * TD * G4 * 4);      // 16.0 MB
  float*  g        = (float*)carve((size_t)B * G4 * 4);           //  0.25 MB
  float*  h_state  = (float*)carve((size_t)B * H * 4);
  float*  c_state  = (float*)carve((size_t)B * H * 4);
  __bf16* hcat_hi  = (__bf16*)carve((size_t)B * 2 * H * 2);
  __bf16* hcat_lo  = (__bf16*)carve((size_t)B * 2 * H * 2);
  __bf16* out_hi   = (__bf16*)carve((size_t)B * H * 2);
  __bf16* out_lo   = (__bf16*)carve((size_t)B * H * 2);
  __bf16* Whh_hi   = (__bf16*)carve((size_t)G4 * H * 2);          //  2.0 MB
  __bf16* Whh_lo   = (__bf16*)carve((size_t)G4 * H * 2);          //  2.0 MB
  __bf16* aW_hi    = (__bf16*)carve((size_t)H * 2 * H * 2);       //  1.0 MB
  __bf16* aW_lo    = (__bf16*)carve((size_t)H * 2 * H * 2);       //  1.0 MB
  __bf16* fcW_hi   = (__bf16*)carve((size_t)V * H * 2);           // 32.8 MB
  __bf16* fcW_lo   = (__bf16*)carve((size_t)V * H * 2);           // 32.8 MB

  const int TB = 256;  // 8 waves/block

  // one-time: weight hi/lo splits (recomputed every call -> deterministic)
  k_split<<<(G4 * H + TB - 1) / TB, TB, 0, stream>>>(W_hh, Whh_hi, Whh_lo, G4 * H);
  k_split<<<(H * 2 * H + TB - 1) / TB, TB, 0, stream>>>(attn_W, aW_hi, aW_lo, H * 2 * H);
  k_split<<<(V * H + TB - 1) / TB, TB, 0, stream>>>(fc_W, fcW_hi, fcW_lo, V * H);

  // init state (+ h0 split into hcat left half)
  k_init<<<(B * H + TB - 1) / TB, TB, 0, stream>>>(h0, c0, h_state, c_state,
                                                   hcat_hi, hcat_lo);

  // embedding + hoisted input-side gate GEMM (f32 WMMA, one-time)
  k_embed<<<(B * TD * E + TB - 1) / TB, TB, 0, stream>>>(dec_X, emb_W, emb);
  {
    int waves = ((B * TD) / 16) * (G4 / 16);    // 16384
    k_xg<<<waves / 8, TB, 0, stream>>>(emb, W_ih, Xg);
  }

  // sequential decode
  for (int t = 0; t < TD; ++t) {
    {
      int waves = 2 * (G4 / 16);                // 256
      k_gates<<<waves / 8, TB, 0, stream>>>(Xg, hcat_hi, hcat_lo,
                                            Whh_hi, Whh_lo, b_ih, b_hh, g, t);
    }
    k_lstm<<<(B * H + TB - 1) / TB, TB, 0, stream>>>(g, c_state, h_state,
                                                     hcat_hi, hcat_lo);
    k_attn<<<B, TE, 0, stream>>>(enc, h_state, hcat_hi, hcat_lo);
    {
      int waves = 2 * (H / 16);                 // 64
      k_attnout<<<waves / 8, TB, 0, stream>>>(hcat_hi, hcat_lo, aW_hi, aW_lo,
                                              attn_b, out_hi, out_lo);
    }
    {
      int waves = 2 * (V / 16);                 // 4000
      k_fc<<<waves / 8, TB, 0, stream>>>(out_hi, out_lo, fcW_hi, fcW_lo,
                                         fc_b, outp, t);
    }
  }

  k_final<<<(B * H + TB - 1) / TB, TB, 0, stream>>>(h_state, c_state, outp);
}
